// CausalSelfAttention_60722247631470
// MI455X (gfx1250) — compile-verified, matrix operand order checked by reading, not run on hardware
//
#include <hip/hip_runtime.h>

// CDNA5 / gfx1250 causal self-attention forward.
// All matrix math via v_wmma_f32_16x16x32_bf16 (f32 accumulate);
// GEMM staging via gfx1250 async global->LDS engine.
// Workspace layout (bytes):
//   xb   @ 0        : 8 MB   x as bf16            [4096][1024]
//   waT  @ 8 MB     : 6 MB   w_attn^T bf16        [3072][1024]
//   wpT  @ 14 MB    : 2 MB   w_proj^T bf16        [1024][1024]
//   qb   @ 16 MB    : 8 MB   Q bf16               [b][h][s][64]
//   kb   @ 24 MB    : 8 MB   K bf16               [b][h][s][64]
//   vtb  @ 32 MB    : 8 MB   V^T bf16             [b][h][64][s]
//   yb   @ 40 MB    : 8 MB   attn out bf16        [b][s][1024]
// Total 48 MB of d_ws.

typedef unsigned short u16;
typedef unsigned int   u32;
typedef __bf16 bf16;
typedef bf16  bf16x16 __attribute__((ext_vector_type(16)));
typedef float f32x8   __attribute__((ext_vector_type(8)));
typedef u32   u32x4   __attribute__((ext_vector_type(4)));
typedef __attribute__((__vector_size__(4 * sizeof(int)))) int i32x4v;
typedef __attribute__((address_space(1))) i32x4v gv_i32x4;
typedef __attribute__((address_space(3))) i32x4v lv_i32x4;

union Frag { bf16x16 v; u32x4 q[2]; };

#if defined(__has_builtin)
#if __has_builtin(__builtin_amdgcn_global_load_async_to_lds_b128)
#define HAVE_ASYNC_LDS 1
#endif
#endif
#ifndef HAVE_ASYNC_LDS
#define HAVE_ASYNC_LDS 0
#endif

__device__ __forceinline__ u16 f2bf(float f) {
  u32 u = __float_as_uint(f);
  u = u + 0x7FFFu + ((u >> 16) & 1u);   // round-to-nearest-even
  return (u16)(u >> 16);
}

// ---------------- conversion kernels ----------------
__global__ void conv_kernel(const float* __restrict__ in, u16* __restrict__ out, int n) {
  int i = blockIdx.x * blockDim.x + threadIdx.x;
  int stride = gridDim.x * blockDim.x;
  for (; i < n; i += stride) out[i] = f2bf(in[i]);
}

// in [K][N] f32 -> out [N][K] bf16
__global__ void transpose_conv_kernel(const float* __restrict__ in, u16* __restrict__ out,
                                      int K, int N) {
  int i = blockIdx.x * blockDim.x + threadIdx.x;
  if (i >= K * N) return;
  int n = i % N, k = i / N;
  out[(size_t)n * K + k] = f2bf(in[i]);
}

// ---------------- bf16 GEMM: C = A[M][K] * Bt[N][K]^T + bias ----------------
// block 128 threads = 4 waves; block tile 128x128; wave tile 64x64 (4x4 WMMA tiles).
#define BK  32
#define LDT 40   // LDS row stride (elems): 80 B, keeps 16B alignment, spreads banks

template<int MODE>  // 0: QKV scatter epilogue (bf16), 1: f32 dense out
__global__ __launch_bounds__(128)
void gemm_kernel(const u16* __restrict__ A, const u16* __restrict__ Bt,
                 const float* __restrict__ bias, float* __restrict__ outf,
                 u16* __restrict__ qb, u16* __restrict__ kb, u16* __restrict__ vtb,
                 int N, int K)
{
  __shared__ __align__(16) u16 As[128 * LDT];
  __shared__ __align__(16) u16 Bs[128 * LDT];
  const int t = threadIdx.x;
  const int lane = t & 31, w = t >> 5;
  const int lr = lane & 15, half = lane >> 4;
  const int m0 = blockIdx.x * 128, n0 = blockIdx.y * 128;
  const int wm = (w >> 1) * 64, wn = (w & 1) * 64;

  const f32x8 vzero = {};
  f32x8 acc[4][4];
#pragma unroll
  for (int i = 0; i < 4; i++)
#pragma unroll
    for (int j = 0; j < 4; j++) acc[i][j] = vzero;

  for (int k0 = 0; k0 < K; k0 += BK) {
#pragma unroll
    for (int i = 0; i < 4; i++) {
      int idx = t + i * 128;            // 512 16B chunks per tile
      int row = idx >> 2, cc = (idx & 3) * 8;
      const u16* ga = A  + (size_t)(m0 + row) * K + k0 + cc;
      const u16* gb = Bt + (size_t)(n0 + row) * K + k0 + cc;
#if HAVE_ASYNC_LDS
      // gfx1250 async-to-LDS engine: global -> LDS without touching VGPR data.
      __builtin_amdgcn_global_load_async_to_lds_b128(
          (gv_i32x4*)ga, (lv_i32x4*)&As[row * LDT + cc], 0, 0);
      __builtin_amdgcn_global_load_async_to_lds_b128(
          (gv_i32x4*)gb, (lv_i32x4*)&Bs[row * LDT + cc], 0, 0);
#else
      *(u32x4*)&As[row * LDT + cc] = *(const u32x4*)ga;
      *(u32x4*)&Bs[row * LDT + cc] = *(const u32x4*)gb;
#endif
    }
    if (k0 + BK < K) {  // prefetch next K-slab into L2
      __builtin_prefetch(A  + (size_t)(m0 + t) * K + k0 + BK, 0, 1);
      __builtin_prefetch(Bt + (size_t)(n0 + t) * K + k0 + BK, 0, 1);
    }
#if HAVE_ASYNC_LDS
    asm volatile("s_wait_asynccnt 0" ::: "memory");
#endif
    __syncthreads();

    Frag a[4], b[4];
#pragma unroll
    for (int mt = 0; mt < 4; mt++) {
      const u16* p = &As[(wm + mt * 16 + lr) * LDT + half * 8];
      a[mt].q[0] = *(const u32x4*)p;          // K = half*8 .. +7
      a[mt].q[1] = *(const u32x4*)(p + 16);   // K = 16+half*8 .. +7
    }
#pragma unroll
    for (int nt = 0; nt < 4; nt++) {
      const u16* p = &Bs[(wn + nt * 16 + lr) * LDT + half * 8];
      b[nt].q[0] = *(const u32x4*)p;
      b[nt].q[1] = *(const u32x4*)(p + 16);
    }
#pragma unroll
    for (int mt = 0; mt < 4; mt++)
#pragma unroll
      for (int nt = 0; nt < 4; nt++)
        acc[mt][nt] = __builtin_amdgcn_wmma_f32_16x16x32_bf16(
            false, a[mt].v, false, b[nt].v, (short)0, acc[mt][nt], false, false);
    __syncthreads();
  }

  // epilogue: C layout VGPR r -> row (r + half*8), col = lane%16
#pragma unroll
  for (int mt = 0; mt < 4; mt++) {
#pragma unroll
    for (int nt = 0; nt < 4; nt++) {
      int col = n0 + wn + nt * 16 + lr;
      float bv = bias[col];
#pragma unroll
      for (int r = 0; r < 8; r++) {
        int row = m0 + wm + mt * 16 + r + half * 8;
        float v = acc[mt][nt][r] + bv;
        if (MODE == 1) {
          outf[(size_t)row * N + col] = v;
        } else {
          int bb = row >> 11, s = row & 2047;
          int which = col >> 10, dd = col & 1023;
          int hh = dd >> 6, c = dd & 63;
          u16 hv = f2bf(v);
          size_t head = (size_t)(bb * 16 + hh);
          if (which == 0)      qb [(head * 2048 + s) * 64 + c] = hv;
          else if (which == 1) kb [(head * 2048 + s) * 64 + c] = hv;
          else                 vtb[(head * 64 + c) * 2048 + s] = hv;   // V transposed
        }
      }
    }
  }
}

// ---------------- flash attention ----------------
// Per-wave 16-query tile: ~170 live VGPRs -> no VGPR-MSB region switching.
// yacc 32 + sacc 32 + Q frags 16 + streamed K/V frags 16 + P frags 16 + stats 16.
#define SLEN 2048
#define NH   16
#define PST  72   // P LDS row stride (elems): 144 B, 16B-aligned rows

template<bool DIAG>
__device__ __forceinline__ void attn_tile(
    int q0, int k0, int lr, int half,
    const u16* __restrict__ kbh, const u16* __restrict__ vtbh,
    const Frag (&aq)[2], f32x8 (&yacc)[4],
    float (&mrow)[8], float (&lrow)[8], u16* pP)
{
  const f32x8 vzero = {};
  f32x8 sacc[4];
#pragma unroll
  for (int nt = 0; nt < 4; nt++) sacc[nt] = vzero;

  // S = Q K^T : stream K fragments one N-tile at a time
#pragma unroll
  for (int nt = 0; nt < 4; nt++) {
    Frag bk[2];
#pragma unroll
    for (int ks = 0; ks < 2; ks++) {
      const u16* p = kbh + (size_t)(k0 + nt * 16 + lr) * 64 + ks * 32 + half * 8;
      bk[ks].q[0] = *(const u32x4*)p;
      bk[ks].q[1] = *(const u32x4*)(p + 16);
    }
#pragma unroll
    for (int ks = 0; ks < 2; ks++)
      sacc[nt] = __builtin_amdgcn_wmma_f32_16x16x32_bf16(
          false, aq[ks].v, false, bk[ks].v, (short)0, sacc[nt], false, false);
  }

  // online softmax (row reductions live inside a 16-lane half: xor 1,2,4,8)
  float rm[8];
#pragma unroll
  for (int r = 0; r < 8; r++) rm[r] = -1e30f;
#pragma unroll
  for (int nt = 0; nt < 4; nt++) {
#pragma unroll
    for (int r = 0; r < 8; r++) {
      float sv = sacc[nt][r] * 0.125f;       // 1/sqrt(64)
      if (DIAG) {
        int key = k0 + nt * 16 + lr;
        int qq  = q0 + r + half * 8;
        if (key > qq) sv = -1e30f;
      }
      sacc[nt][r] = sv;
      rm[r] = fmaxf(rm[r], sv);
    }
  }
#pragma unroll
  for (int r = 0; r < 8; r++) {
#pragma unroll
    for (int msk = 1; msk < 16; msk <<= 1)
      rm[r] = fmaxf(rm[r], __shfl_xor(rm[r], msk, 32));
    float mnew  = fmaxf(mrow[r], rm[r]);
    float alpha = __expf(mrow[r] - mnew);
    mrow[r] = mnew;
    lrow[r] *= alpha;
#pragma unroll
    for (int ht = 0; ht < 4; ht++) yacc[ht][r] *= alpha;
  }
  float rs[8];
#pragma unroll
  for (int r = 0; r < 8; r++) rs[r] = 0.0f;
#pragma unroll
  for (int nt = 0; nt < 4; nt++) {
#pragma unroll
    for (int r = 0; r < 8; r++) {
      float p = __expf(sacc[nt][r] - mrow[r]);
      rs[r] += p;
      pP[(r + half * 8) * PST + nt * 16 + lr] = f2bf(p);
    }
  }
#pragma unroll
  for (int r = 0; r < 8; r++) {
#pragma unroll
    for (int msk = 1; msk < 16; msk <<= 1)
      rs[r] += __shfl_xor(rs[r], msk, 32);
    lrow[r] += rs[r];
  }
  asm volatile("s_wait_dscnt 0" ::: "memory");   // P stores visible before frag reads
  __builtin_amdgcn_wave_barrier();

  // P fragments (A layout) from LDS
  Frag ap[2];
#pragma unroll
  for (int ks = 0; ks < 2; ks++) {
    const u16* p = &pP[lr * PST + ks * 32 + half * 8];
    ap[ks].q[0] = *(const u32x4*)p;
    ap[ks].q[1] = *(const u32x4*)(p + 16);
  }

  // y += P V^T : stream V fragments one H-tile at a time
#pragma unroll
  for (int ht = 0; ht < 4; ht++) {
    Frag bv[2];
#pragma unroll
    for (int ks = 0; ks < 2; ks++) {
      const u16* p = vtbh + (size_t)(ht * 16 + lr) * SLEN + k0 + ks * 32 + half * 8;
      bv[ks].q[0] = *(const u32x4*)p;
      bv[ks].q[1] = *(const u32x4*)(p + 16);
    }
#pragma unroll
    for (int ks = 0; ks < 2; ks++)
      yacc[ht] = __builtin_amdgcn_wmma_f32_16x16x32_bf16(
          false, ap[ks].v, false, bv[ks].v, (short)0, yacc[ht], false, false);
  }
  asm volatile("s_wait_dscnt 0" ::: "memory");   // WAR: frag reads before next P stores
  __builtin_amdgcn_wave_barrier();
}

__global__ __launch_bounds__(256)
void attn_kernel(const u16* __restrict__ qb, const u16* __restrict__ kb,
                 const u16* __restrict__ vtb, u16* __restrict__ yb)
{
  __shared__ __align__(16) u16 Ps[8][16 * PST];
  const int t = threadIdx.x;
  const int lane = t & 31, w = t >> 5;
  const int lr = lane & 15, half = lane >> 4;
  const int gw = blockIdx.x * 8 + w;        // 4096 waves total
  const int qt = gw & 127;                  // query tile (of 16)
  const int hh = (gw >> 7) & 15;
  const int bb = gw >> 11;
  const int q0 = qt * 16;
  const u16* kbh  = kb  + (size_t)(bb * NH + hh) * SLEN * 64;
  const u16* qbh  = qb  + (size_t)(bb * NH + hh) * SLEN * 64;
  const u16* vtbh = vtb + (size_t)(bb * NH + hh) * 64 * SLEN;
  u16* pP = &Ps[w][0];

  // Q fragments (reduction dim hd contiguous), loaded once
  Frag aq[2];
#pragma unroll
  for (int ks = 0; ks < 2; ks++) {
    const u16* p = qbh + (size_t)(q0 + lr) * 64 + ks * 32 + half * 8;
    aq[ks].q[0] = *(const u32x4*)p;
    aq[ks].q[1] = *(const u32x4*)(p + 16);
  }

  const f32x8 vzero = {};
  f32x8 yacc[4];
  float mrow[8], lrow[8];
#pragma unroll
  for (int ht = 0; ht < 4; ht++) yacc[ht] = vzero;
#pragma unroll
  for (int r = 0; r < 8; r++) { mrow[r] = -1e30f; lrow[r] = 0.0f; }

  // causal: full (unmasked) 64-key tiles, then one masked diagonal tile
  const int nf = q0 >> 6;
  for (int kt = 0; kt < nf; kt++) {
    const int k0n = (kt + 1) * 64;          // prefetch next tile into L2
    __builtin_prefetch(kbh + (size_t)(k0n + lane) * 64, 0, 1);
    __builtin_prefetch(kbh + (size_t)(k0n + 32 + lane) * 64, 0, 1);
    __builtin_prefetch(vtbh + (size_t)lane * SLEN + k0n, 0, 1);
    __builtin_prefetch(vtbh + (size_t)(lane + 32) * SLEN + k0n, 0, 1);
    attn_tile<false>(q0, kt * 64, lr, half, kbh, vtbh, aq, yacc, mrow, lrow, pP);
  }
  attn_tile<true>(q0, nf * 64, lr, half, kbh, vtbh, aq, yacc, mrow, lrow, pP);

  // normalize, write y bf16 [b][s][1024]
#pragma unroll
  for (int r = 0; r < 8; r++) {
    float inv = 1.0f / lrow[r];
    int qq = q0 + r + half * 8;
    size_t base = ((size_t)bb * SLEN + qq) * 1024 + hh * 64;
#pragma unroll
    for (int ht = 0; ht < 4; ht++)
      yb[base + ht * 16 + lr] = f2bf(yacc[ht][r] * inv);
  }
}

// ---------------- launch ----------------
extern "C" void kernel_launch(void* const* d_in, const int* in_sizes, int n_in,
                              void* d_out, int out_size, void* d_ws, size_t ws_size,
                              hipStream_t stream) {
  const float* x      = (const float*)d_in[0];
  const float* w_attn = (const float*)d_in[1];
  const float* b_attn = (const float*)d_in[2];
  const float* w_proj = (const float*)d_in[3];
  const float* b_proj = (const float*)d_in[4];
  float* out = (float*)d_out;

  char* ws  = (char*)d_ws;
  u16* xb  = (u16*)(ws);
  u16* waT = (u16*)(ws + (size_t)8  * 1024 * 1024);
  u16* wpT = (u16*)(ws + (size_t)14 * 1024 * 1024);
  u16* qbf = (u16*)(ws + (size_t)16 * 1024 * 1024);
  u16* kbf = (u16*)(ws + (size_t)24 * 1024 * 1024);
  u16* vtb = (u16*)(ws + (size_t)32 * 1024 * 1024);
  u16* yb  = (u16*)(ws + (size_t)40 * 1024 * 1024);

  // 1) precision conversion (+ weight transpose so both GEMM operands are [row][K])
  conv_kernel<<<2048, 256, 0, stream>>>(x, xb, 4096 * 1024);
  transpose_conv_kernel<<<(1024 * 3072) / 256, 256, 0, stream>>>(w_attn, waT, 1024, 3072);
  transpose_conv_kernel<<<(1024 * 1024) / 256, 256, 0, stream>>>(w_proj, wpT, 1024, 1024);

  // 2) QKV GEMM: [4096][1024] x [1024][3072] -> scatter Q/K/V^T per head (bf16)
  gemm_kernel<0><<<dim3(32, 24), 128, 0, stream>>>(xb, waT, b_attn, nullptr,
                                                   qbf, kbf, vtb, 3072, 1024);

  // 3) causal flash attention (WMMA bf16, f32 softmax)
  attn_kernel<<<512, 256, 0, stream>>>(qbf, kbf, vtb, yb);

  // 4) output projection: [4096][1024] x [1024][1024] + bias -> f32 out
  gemm_kernel<1><<<dim3(32, 8), 128, 0, stream>>>(yb, wpT, b_proj, out,
                                                  nullptr, nullptr, nullptr, 1024, 1024);
}
